// LinearSelfAttention_9758165697169
// MI455X (gfx1250) — matrix operand order repaired and checked by reading, not compiled, not run on hardware
//
#include <hip/hip_runtime.h>

#define DMODEL 512
#define NHEADS 8
#define HDIM   64
#define BATCH  4
#define SEQ    8192
#define BN     (BATCH * SEQ)        // 32768
#define EPSV   1e-6f

typedef __attribute__((ext_vector_type(16))) __bf16 v16bf;
typedef __attribute__((ext_vector_type(8)))  __bf16 v8bf;
typedef __attribute__((ext_vector_type(8)))  float  v8f;

__device__ __forceinline__ v16bf pack16(v8bf lo, v8bf hi) {
  v16bf r;
#pragma unroll
  for (int i = 0; i < 8; ++i) { r[i] = lo[i]; r[i + 8] = hi[i]; }
  return r;
}

// A-fragment (16x32 bf16, contiguous-K LDS tile, row stride `ldm` elems)
__device__ __forceinline__ v16bf load_a_contig(const __bf16* lds, int row, int ldm,
                                               int kt, int half) {
  const __bf16* base = lds + row * ldm + kt;
  v8bf lo = *(const v8bf*)(base + half * 16);
  v8bf hi = *(const v8bf*)(base + half * 16 + 8);
  return pack16(lo, hi);
}

// B-fragment (32x16 bf16) from col-major LDS tile [n][k], row stride ldm elems
__device__ __forceinline__ v16bf load_b_nk(const __bf16* lds, int col, int ldm,
                                           int kt, int half) {
  const __bf16* base = lds + col * ldm + kt + half * 16;
  v8bf lo = *(const v8bf*)(base);
  v8bf hi = *(const v8bf*)(base + 8);
  return pack16(lo, hi);
}

// ---- gfx1250 async global->LDS copy (ASYNCcnt path, §10 VMEM) -------------
__device__ __forceinline__ unsigned lds_off_of(const void* p) {
  // flat LDS-aperture address: low 32 bits are the LDS byte offset
  return (unsigned)(unsigned long long)p;
}
__device__ __forceinline__ void async_copy_b128(unsigned lds_byte_off,
                                                const void* gbase,
                                                unsigned gbyte_off) {
  asm volatile("global_load_async_to_lds_b128 %0, %1, %2"
               :
               : "v"(lds_byte_off), "v"(gbyte_off), "s"(gbase)
               : "memory");
}
__device__ __forceinline__ void wait_async() {
  asm volatile("s_wait_asynccnt 0" ::: "memory");
}

// ---------------------------------------------------------------------------
// Kernel 1: fused QKV projection  out = featuremap(x @ W^T + b)
// grid = (BN/128, DMODEL/64, 3)   block = 128 (4 waves)
// wave w computes rows [w*32, w*32+32) x 64 cols : 2 A-frags, 4 B-frags, 8 WMMA
// ---------------------------------------------------------------------------
__global__ void __launch_bounds__(128) qkv_gemm(
    const float* __restrict__ x,
    const float* __restrict__ Wq, const float* __restrict__ biasq,
    const float* __restrict__ Wk, const float* __restrict__ biask,
    const float* __restrict__ Wv, const float* __restrict__ biasv,
    __bf16* __restrict__ qo, __bf16* __restrict__ ko, __bf16* __restrict__ vo) {
  const int which = blockIdx.z;
  const float* W    = (which == 0) ? Wq : (which == 1) ? Wk : Wv;
  const float* bias = (which == 0) ? biasq : (which == 1) ? biask : biasv;
  __bf16* out       = (which == 0) ? qo : (which == 1) ? ko : vo;
  const bool fmap = (which < 2);

  const int m0 = blockIdx.x * 128;
  const int n0 = blockIdx.y * 64;
  const int t = threadIdx.x, lane = t & 31, wid = t >> 5, half = (lane >> 4) & 1;

  __shared__ __attribute__((aligned(16))) __bf16 lds_x[128 * 32];
  __shared__ __attribute__((aligned(16))) __bf16 lds_w[64 * 32];

  v8f acc[2][4] = {};

  for (int kt = 0; kt < DMODEL; kt += 32) {
#pragma unroll
    for (int i = 0; i < 8; ++i) {               // 1024 float4 chunks of X tile
      int c = t + i * 128;
      int row = c >> 3;
      int kc = (c & 7) * 4;
      float4 xf = *(const float4*)(x + (size_t)(m0 + row) * DMODEL + kt + kc);
      lds_x[row * 32 + kc + 0] = (__bf16)xf.x;
      lds_x[row * 32 + kc + 1] = (__bf16)xf.y;
      lds_x[row * 32 + kc + 2] = (__bf16)xf.z;
      lds_x[row * 32 + kc + 3] = (__bf16)xf.w;
    }
#pragma unroll
    for (int i = 0; i < 4; ++i) {               // 512 float4 chunks of W tile
      int c = t + i * 128;
      int row = c >> 3;
      int kc = (c & 7) * 4;
      float4 wf = *(const float4*)(W + (size_t)(n0 + row) * DMODEL + kt + kc);
      lds_w[row * 32 + kc + 0] = (__bf16)wf.x;
      lds_w[row * 32 + kc + 1] = (__bf16)wf.y;
      lds_w[row * 32 + kc + 2] = (__bf16)wf.z;
      lds_w[row * 32 + kc + 3] = (__bf16)wf.w;
    }
    __syncthreads();

    v16bf a0 = load_a_contig(lds_x, wid * 32 + (lane & 15), 32, 0, half);
    v16bf a1 = load_a_contig(lds_x, wid * 32 + 16 + (lane & 15), 32, 0, half);
#pragma unroll
    for (int j = 0; j < 4; ++j) {
      int bcol = j * 16 + (lane & 15);
      v16bf b = load_b_nk(lds_w, bcol, 32, 0, half);
      acc[0][j] = __builtin_amdgcn_wmma_f32_16x16x32_bf16(
          false, a0, false, b, (short)0, acc[0][j], false, false);
      acc[1][j] = __builtin_amdgcn_wmma_f32_16x16x32_bf16(
          false, a1, false, b, (short)0, acc[1][j], false, false);
    }
    __syncthreads();
  }

  // epilogue: bias + feature map, scatter to [B,H,N,D] bf16
#pragma unroll
  for (int mi = 0; mi < 2; ++mi) {
#pragma unroll
    for (int j = 0; j < 4; ++j) {
      int n = n0 + j * 16 + (lane & 15);
      float bb = bias[n];
      int h = n >> 6, d = n & 63;
#pragma unroll
      for (int r = 0; r < 8; ++r) {
        int m = m0 + wid * 32 + mi * 16 + half * 8 + r;
        float val = acc[mi][j][r] + bb;
        if (fmap) val = (val > 0.f) ? (val + 1.f) : __expf(val);   // elu(x)+1
        int bidx = m >> 13;        // / SEQ
        int tok = m & (SEQ - 1);
        out[((size_t)(bidx * NHEADS + h) * SEQ + tok) * HDIM + d] = (__bf16)val;
      }
    }
  }
}

// ---------------------------------------------------------------------------
// Kernel 2: k_sum[bh][d] = sum_n k[bh][n][d]
// ---------------------------------------------------------------------------
__global__ void __launch_bounds__(64) ksum_kernel(const __bf16* __restrict__ k,
                                                  float* __restrict__ ksum) {
  int bh = blockIdx.x, d = threadIdx.x;
  const __bf16* p = k + (size_t)bh * SEQ * HDIM + d;
  float s = 0.f;
  for (int n = 0; n < SEQ; ++n) s += (float)p[(size_t)n * HDIM];
  ksum[bh * HDIM + d] = s;
}

// ---------------------------------------------------------------------------
// Kernel 3: kv[bh] = K^T V  (64x64, f32 accum).  grid = B*H, block = 128.
// Token tiles streamed global->LDS with async copies (ASYNCcnt).
// ---------------------------------------------------------------------------
__global__ void __launch_bounds__(128) kv_kernel(const __bf16* __restrict__ k,
                                                 const __bf16* __restrict__ v,
                                                 float* __restrict__ kv) {
  const int bh = blockIdx.x;
  const int t = threadIdx.x, lane = t & 31, wid = t >> 5, half = (lane >> 4) & 1;
  __shared__ __attribute__((aligned(16))) __bf16 lds_k[32 * 64];
  __shared__ __attribute__((aligned(16))) __bf16 lds_v[32 * 64];
  v8f acc[4] = {};
  const __bf16* kb = k + (size_t)bh * SEQ * HDIM;
  const __bf16* vb = v + (size_t)bh * SEQ * HDIM;
  const unsigned lk = lds_off_of(lds_k), lv = lds_off_of(lds_v);

  for (int n0 = 0; n0 < SEQ; n0 += 32) {
#pragma unroll
    for (int i = 0; i < 2; ++i) {
      int c = t + i * 128;                  // 256 x 16B chunks per matrix
      int tok = c >> 3;
      int off = (c & 7) * 8;
      unsigned loff = (unsigned)(tok * 64 + off) * 2u;
      unsigned goff = (unsigned)((n0 + tok) * HDIM + off) * 2u;
      async_copy_b128(lk + loff, kb, goff);
      async_copy_b128(lv + loff, vb, goff);
    }
    wait_async();
    __syncthreads();

    // A = K^T : row = d1, K-dim = token (transposed gather from [tok][d] tile)
    const int arow = wid * 16 + (lane & 15);
    v16bf a;
#pragma unroll
    for (int e = 0; e < 16; ++e) {
      int kk = half * 16 + e;
      a[e] = lds_k[kk * 64 + arow];
    }
#pragma unroll
    for (int j = 0; j < 4; ++j) {
      int bcol = j * 16 + (lane & 15);
      v16bf b;
#pragma unroll
      for (int e = 0; e < 16; ++e) {
        int kk = half * 16 + e;
        b[e] = lds_v[kk * 64 + bcol];
      }
      acc[j] = __builtin_amdgcn_wmma_f32_16x16x32_bf16(false, a, false, b,
                                                       (short)0, acc[j], false, false);
    }
    __syncthreads();
  }

#pragma unroll
  for (int j = 0; j < 4; ++j) {
    int d2 = j * 16 + (lane & 15);
#pragma unroll
    for (int r = 0; r < 8; ++r) {
      int d1 = wid * 16 + half * 8 + r;
      kv[((size_t)bh * HDIM + d1) * HDIM + d2] = acc[j][r];
    }
  }
}

// ---------------------------------------------------------------------------
// Kernel 4: y = (Q @ kv) / (Q . k_sum + eps), write bf16 [B,N,C] layout
// grid = (SEQ/64, B*H), block = 128;  q tile fetched with async copies
// ---------------------------------------------------------------------------
__global__ void __launch_bounds__(128) attn_out_kernel(
    const __bf16* __restrict__ q, const float* __restrict__ kv,
    const float* __restrict__ ksum, __bf16* __restrict__ y) {
  const int bh = blockIdx.y;
  const int n0 = blockIdx.x * 64;
  const int bidx = bh / NHEADS, h = bh % NHEADS;
  const int t = threadIdx.x, lane = t & 31, wid = t >> 5, half = (lane >> 4) & 1;

  __shared__ __attribute__((aligned(16))) __bf16 lds_q[64 * 64];
  __shared__ __attribute__((aligned(16))) __bf16 lds_kv[64 * 64];
  __shared__ float lds_ks[64];
  __shared__ float lds_z[64];

  const __bf16* qb = q + (size_t)bh * SEQ * HDIM;
  const unsigned lq = lds_off_of(lds_q);
#pragma unroll
  for (int i = 0; i < 4; ++i) {
    int c = t + i * 128;                    // 512 x 16B chunks
    int tok = c >> 3;
    int off = (c & 7) * 8;
    async_copy_b128(lq + (unsigned)(tok * 64 + off) * 2u, qb,
                    (unsigned)((n0 + tok) * HDIM + off) * 2u);
  }
  const float* kvb = kv + (size_t)bh * HDIM * HDIM;
#pragma unroll
  for (int i = 0; i < 32; ++i) {
    int c = t + i * 128;
    lds_kv[c] = (__bf16)kvb[c];
  }
  if (t < 64) lds_ks[t] = ksum[bh * HDIM + t];
  wait_async();
  __syncthreads();

  // z per token row
  if (t < 64) {
    float s = EPSV;
#pragma unroll 8
    for (int d = 0; d < HDIM; ++d) s += (float)lds_q[t * 64 + d] * lds_ks[d];
    lds_z[t] = s;
  }
  __syncthreads();

  v8f acc[4] = {};
  const int arow = wid * 16 + (lane & 15);
#pragma unroll
  for (int kt = 0; kt < HDIM; kt += 32) {
    v16bf a = load_a_contig(lds_q, arow, 64, kt, half);
#pragma unroll
    for (int j = 0; j < 4; ++j) {
      int bcol = j * 16 + (lane & 15);
      v16bf b;
#pragma unroll
      for (int e = 0; e < 16; ++e) {
        int kk = kt + half * 16 + e;
        b[e] = lds_kv[kk * 64 + bcol];
      }
      acc[j] = __builtin_amdgcn_wmma_f32_16x16x32_bf16(false, a, false, b,
                                                       (short)0, acc[j], false, false);
    }
  }

#pragma unroll
  for (int j = 0; j < 4; ++j) {
    int d2 = j * 16 + (lane & 15);
#pragma unroll
    for (int r = 0; r < 8; ++r) {
      int rl = wid * 16 + half * 8 + r;
      int tok = n0 + rl;
      float val = acc[j][r] / lds_z[rl];
      y[((size_t)bidx * SEQ + tok) * DMODEL + h * HDIM + d2] = (__bf16)val;
    }
  }
}

// ---------------------------------------------------------------------------
// Kernel 5: out = Y @ Wo^T + bo  (f32 output)
// grid = (BN/128, DMODEL/64), block = 128; Y tile fetched with async copies
// ---------------------------------------------------------------------------
__global__ void __launch_bounds__(128) out_gemm(const __bf16* __restrict__ y,
                                                const float* __restrict__ Wo,
                                                const float* __restrict__ bo,
                                                float* __restrict__ out) {
  const int m0 = blockIdx.x * 128;
  const int n0 = blockIdx.y * 64;
  const int t = threadIdx.x, lane = t & 31, wid = t >> 5, half = (lane >> 4) & 1;

  __shared__ __attribute__((aligned(16))) __bf16 lds_y[128 * 32];
  __shared__ __attribute__((aligned(16))) __bf16 lds_w[64 * 32];
  const unsigned ly = lds_off_of(lds_y);

  v8f acc[2][4] = {};

  for (int kt = 0; kt < DMODEL; kt += 32) {
#pragma unroll
    for (int i = 0; i < 4; ++i) {
      int c = t + i * 128;                  // 512 x 16B chunks of Y tile
      int row = c >> 2;
      int off = (c & 3) * 8;
      async_copy_b128(ly + (unsigned)(row * 32 + off) * 2u, y,
                      (unsigned)((m0 + row) * DMODEL + kt + off) * 2u);
    }
#pragma unroll
    for (int i = 0; i < 4; ++i) {
      int c = t + i * 128;
      int row = c >> 3;
      int kc = (c & 7) * 4;
      float4 wf = *(const float4*)(Wo + (size_t)(n0 + row) * DMODEL + kt + kc);
      lds_w[row * 32 + kc + 0] = (__bf16)wf.x;
      lds_w[row * 32 + kc + 1] = (__bf16)wf.y;
      lds_w[row * 32 + kc + 2] = (__bf16)wf.z;
      lds_w[row * 32 + kc + 3] = (__bf16)wf.w;
    }
    wait_async();
    __syncthreads();

    v16bf a0 = load_a_contig(lds_y, wid * 32 + (lane & 15), 32, 0, half);
    v16bf a1 = load_a_contig(lds_y, wid * 32 + 16 + (lane & 15), 32, 0, half);
#pragma unroll
    for (int j = 0; j < 4; ++j) {
      int bcol = j * 16 + (lane & 15);
      v16bf b = load_b_nk(lds_w, bcol, 32, 0, half);
      acc[0][j] = __builtin_amdgcn_wmma_f32_16x16x32_bf16(
          false, a0, false, b, (short)0, acc[0][j], false, false);
      acc[1][j] = __builtin_amdgcn_wmma_f32_16x16x32_bf16(
          false, a1, false, b, (short)0, acc[1][j], false, false);
    }
    __syncthreads();
  }

#pragma unroll
  for (int mi = 0; mi < 2; ++mi) {
#pragma unroll
    for (int j = 0; j < 4; ++j) {
      int n = n0 + j * 16 + (lane & 15);
      float bb = bo[n];
#pragma unroll
      for (int r = 0; r < 8; ++r) {
        int m = m0 + wid * 32 + mi * 16 + half * 8 + r;
        out[(size_t)m * DMODEL + n] = acc[mi][j][r] + bb;
      }
    }
  }
}

// ---------------------------------------------------------------------------
extern "C" void kernel_launch(void* const* d_in, const int* in_sizes, int n_in,
                              void* d_out, int out_size, void* d_ws, size_t ws_size,
                              hipStream_t stream) {
  const float* x  = (const float*)d_in[0];
  const float* Wq = (const float*)d_in[1];
  const float* bq = (const float*)d_in[2];
  const float* Wk = (const float*)d_in[3];
  const float* bk = (const float*)d_in[4];
  const float* Wv = (const float*)d_in[5];
  const float* bv = (const float*)d_in[6];
  const float* Wo = (const float*)d_in[7];
  const float* bo = (const float*)d_in[8];
  float* out = (float*)d_out;

  char* ws = (char*)d_ws;
  const size_t szQ = (size_t)BN * DMODEL * sizeof(__bf16);   // 32 MB each
  __bf16* q = (__bf16*)(ws);
  __bf16* k = (__bf16*)(ws + szQ);
  __bf16* v = (__bf16*)(ws + 2 * szQ);
  __bf16* y = (__bf16*)(ws + 3 * szQ);
  float* kvbuf = (float*)(ws + 4 * szQ);
  float* ksum  = (float*)(ws + 4 * szQ +
                          (size_t)BATCH * NHEADS * HDIM * HDIM * sizeof(float));

  qkv_gemm<<<dim3(BN / 128, DMODEL / 64, 3), 128, 0, stream>>>(
      x, Wq, bq, Wk, bk, Wv, bv, q, k, v);
  ksum_kernel<<<dim3(BATCH * NHEADS), 64, 0, stream>>>(k, ksum);
  kv_kernel<<<dim3(BATCH * NHEADS), 128, 0, stream>>>(k, v, kvbuf);
  attn_out_kernel<<<dim3(SEQ / 64, BATCH * NHEADS), 128, 0, stream>>>(
      q, kvbuf, ksum, y);
  out_gemm<<<dim3(BN / 128, DMODEL / 64), 128, 0, stream>>>(y, Wo, bo, out);
}
